// DepthwiseXCorr_86887188398983
// MI455X (gfx1250) — matrix operand order repaired and checked by reading, not compiled, not run on hardware
//
#include <hip/hip_runtime.h>
#include <hip/hip_bf16.h>

typedef __attribute__((ext_vector_type(16))) _Float16 v16h;
typedef __attribute__((ext_vector_type(8)))  _Float16 v8h;
typedef __attribute__((ext_vector_type(8)))  float    v8f;
typedef __attribute__((ext_vector_type(4)))  unsigned int v4u;
typedef __attribute__((ext_vector_type(8)))  int v8i;
typedef __attribute__((ext_vector_type(4)))  int v4i;

#define EPSV 1e-5f

#if __has_builtin(__builtin_amdgcn_tensor_load_to_lds)
#define USE_TDM 1
#else
#define USE_TDM 0
#endif

// ---------------------------------------------------------------------------
// Weight packing: f32 -> f16 (row-major [O][K]).
// ---------------------------------------------------------------------------
__global__ void pack_f16_kernel(const float* __restrict__ src,
                                _Float16* __restrict__ dst, int n) {
  int i = blockIdx.x * 256 + threadIdx.x;
  if (i < n) dst[i] = (_Float16)src[i];
}

// h2_w is [10,256]; pad to [16,256] with zeros so the WMMA tile is full.
__global__ void pack_h2_kernel(const float* __restrict__ src,
                               _Float16* __restrict__ dst) {
  int i = blockIdx.x * 256 + threadIdx.x;     // 0 .. 16*256-1
  int o = i >> 8, k = i & 255;
  dst[i] = (o < 10) ? (_Float16)src[o * 256 + k] : (_Float16)0.0f;
}

// Fold eval-BN into per-channel scale/shift so hot epilogues are fma+max only.
__global__ void bn_prep_kernel(const float* __restrict__ g, const float* __restrict__ b,
                               const float* __restrict__ m, const float* __restrict__ v,
                               float* __restrict__ sc, float* __restrict__ sh) {
  int i = threadIdx.x;                        // 256 channels
  float s = g[i] / sqrtf(v[i] + EPSV);
  sc[i] = s;
  sh[i] = b[i] - m[i] * s;
}

// ---------------------------------------------------------------------------
// 3x3 VALID conv + BN + ReLU as implicit GEMM via v_wmma_f32_16x16x32_f16.
//   M = out-channels, K = 2304, N = B*Ho*Wo pixels (divides by 16 exactly).
// Each wave: ONE im2col B fragment, FOUR out-channel tiles -> 4 independent
// WMMAs per k-step (pipelines the XDL) and 4x amortized gather cost.
// Block = 128 threads = 4 waves = 4 pixel tiles. grid.y = 4 otile groups.
// ---------------------------------------------------------------------------
__global__ __launch_bounds__(128)
void conv3_bn_relu_wmma(const float* __restrict__ in,
                        const _Float16* __restrict__ wf16,  // [256][2304]
                        const float* __restrict__ sc256, const float* __restrict__ sh256,
                        _Float16* __restrict__ out,
                        int H, int W, int Ho, int Wo) {
  const int lane  = threadIdx.x & 31;
  const int wv    = threadIdx.x >> 5;
  const int ptile = blockIdx.x * 4 + wv;
  const int og    = blockIdx.y;              // otiles og*4 .. og*4+3
  const int n  = lane & 15;
  const int hi = lane >> 4;
  const int kbA = hi * 16;
  const int kbB = hi * 16;
  const int NPIX = Ho * Wo;

  const int gpix = ptile * 16 + n;
  const int bt = gpix / NPIX;
  const int p  = gpix - bt * NPIX;
  const int y  = p / Wo;
  const int x  = p - y * Wo;

  const float* inb = in + (size_t)bt * 256 * H * W;
  const _Float16* wrow0 = wf16 + (size_t)((og * 4 + 0) * 16 + n) * 2304;
  const _Float16* wrow1 = wrow0 + (size_t)16 * 2304;
  const _Float16* wrow2 = wrow1 + (size_t)16 * 2304;
  const _Float16* wrow3 = wrow2 + (size_t)16 * 2304;

  v8f acc0 = {}, acc1 = {}, acc2 = {}, acc3 = {};
  for (int kc = 0; kc < 72; ++kc) {            // K = 2304 = 72 * 32
    const int k0 = kc * 32;
    // ---- shared B fragment (im2col gather, f32 -> f16): K = k0 + kbB + h ----
    int kk = k0 + kbB;
    int c  = kk / 9;
    int rem = kk - c * 9;
    v16h bf;
    #pragma unroll
    for (int h = 0; h < 16; ++h) {
      int r  = rem / 3;
      int s2 = rem - r * 3;
      bf[h] = (_Float16)inb[((size_t)c * H + (y + r)) * W + (x + s2)];
      if (++rem == 9) { rem = 0; ++c; }
    }
    // ---- 4 A fragments (vector loads) + 4 independent WMMAs ----
    #define LOAD_A(av, wr)                                                 \
      { v8h alo = *(const v8h*)((wr) + k0 + kbA);                          \
        v8h ahi = *(const v8h*)((wr) + k0 + kbA + 8);                      \
        _Pragma("unroll")                                                  \
        for (int i = 0; i < 8; ++i) { av[i] = alo[i]; av[8 + i] = ahi[i]; } }
    v16h a0, a1, a2, a3;
    LOAD_A(a0, wrow0) LOAD_A(a1, wrow1) LOAD_A(a2, wrow2) LOAD_A(a3, wrow3)
    #undef LOAD_A
    acc0 = __builtin_amdgcn_wmma_f32_16x16x32_f16(false, a0, false, bf, (short)0, acc0, false, false);
    acc1 = __builtin_amdgcn_wmma_f32_16x16x32_f16(false, a1, false, bf, (short)0, acc1, false, false);
    acc2 = __builtin_amdgcn_wmma_f32_16x16x32_f16(false, a2, false, bf, (short)0, acc2, false, false);
    acc3 = __builtin_amdgcn_wmma_f32_16x16x32_f16(false, a3, false, bf, (short)0, acc3, false, false);
  }
  // ---- epilogue: BN (prefolded) + ReLU, store f16 ----
  #pragma unroll
  for (int ot = 0; ot < 4; ++ot) {
    const v8f acc = (ot == 0) ? acc0 : (ot == 1) ? acc1 : (ot == 2) ? acc2 : acc3;
    #pragma unroll
    for (int r = 0; r < 8; ++r) {
      int o = (og * 4 + ot) * 16 + 8 * hi + r;
      float val = fmaxf(acc[r] * sc256[o] + sh256[o], 0.0f);
      out[((size_t)(bt * 256 + o) * Ho + y) * Wo + x] = (_Float16)val;
    }
  }
}

// ---------------------------------------------------------------------------
// Depthwise cross-correlation: per (b,c), 29x29 (x) 5x5 -> 25x25.
// Search tile (1682 B) staged into LDS via the Tensor Data Mover (wave 0
// issues, s_wait_tensorcnt, barrier). feat written [pixel][channel] f16 so
// head B-fragments are single v16h loads.
// ---------------------------------------------------------------------------
__global__ __launch_bounds__(128)
void xcorr_dw_kernel(const _Float16* __restrict__ s,
                     const _Float16* __restrict__ k,
                     _Float16* __restrict__ feat) {
  __shared__ __align__(16) _Float16 sSh[848];
  __shared__ float sS[841];
  __shared__ float sK[25];
  const int bc = blockIdx.x;                 // b*256 + c
  const _Float16* sp = s + (size_t)bc * 841;
  const _Float16* kp = k + (size_t)bc * 25;

#if USE_TDM
  if (threadIdx.x < 32) {                    // one wave issues the DMA
    unsigned lds_off = (unsigned)(uintptr_t)(&sSh[0]);   // low 32b = LDS offset
    unsigned long long ga = (unsigned long long)(uintptr_t)sp;
    v4u g0;
    g0[0] = 1u;                                          // count=1, load, user
    g0[1] = lds_off;                                     // lds_addr
    g0[2] = (unsigned)(ga & 0xffffffffu);                // global_addr[31:0]
    g0[3] = (unsigned)((ga >> 32) & 0x1ffffffu) | (2u << 30); // [56:32] | type=2
    v8i g1;
    g1[0] = (1 << 16);        // workgroup_mask=0, data_size=1 (2B elements)
    g1[1] = (841 << 16);      // tensor_dim0[15:0] (<<16 past atomic_barrier_addr)
    g1[2] = (1 << 16);        // tensor_dim0[31:16]=0 | tensor_dim1[15:0]=1
    g1[3] = (841 << 16);      // tensor_dim1[31:16]=0 | tile_dim0=841
    g1[4] = 1;                // tile_dim1=1, tile_dim2=0
    g1[5] = 841;              // tensor_dim0_stride[31:0]
    g1[6] = 0;                // stride[47:32]=0 | tensor_dim1_stride[15:0]=0
    g1[7] = 0;
    v4i gz4 = {0, 0, 0, 0};
    v8i gz8 = {0, 0, 0, 0, 0, 0, 0, 0};
    __builtin_amdgcn_tensor_load_to_lds(g0, g1, gz4, gz4, gz8, 0);
    __builtin_amdgcn_s_wait_tensorcnt(0);
  }
  __syncthreads();
  for (int i = threadIdx.x; i < 841; i += 128) sS[i] = (float)sSh[i];
#else
  for (int i = threadIdx.x; i < 841; i += 128) sS[i] = (float)sp[i];
#endif
  if (threadIdx.x < 25) sK[threadIdx.x] = (float)kp[threadIdx.x];
  __syncthreads();

  const int bt = bc >> 8, c = bc & 255;
  for (int t = threadIdx.x; t < 625; t += 128) {
    int y = t / 25, x = t - y * 25;
    float a = 0.0f;
    #pragma unroll
    for (int r = 0; r < 5; ++r)
      #pragma unroll
      for (int s2 = 0; s2 < 5; ++s2)
        a += sS[(y + r) * 29 + (x + s2)] * sK[r * 5 + s2];
    feat[(size_t)(bt * 625 + t) * 256 + c] = (_Float16)a;
  }
}

// ---------------------------------------------------------------------------
// Fused head: x = relu(bn(h1_w @ feat)); out = h2_w @ x + h2_b.
// One block per 16-pixel tile (2500 blocks). 16 waves each compute one 16-row
// tile of x (8 WMMAs, K=256), stash f16 into LDS [col][c], barrier, wave 0
// computes the 16x16 (10 valid rows) output tile (8 WMMAs).
// ---------------------------------------------------------------------------
__global__ __launch_bounds__(512)
void head_wmma_kernel(const _Float16* __restrict__ feat,   // [40000][256]
                      const _Float16* __restrict__ w1,     // [256][256]
                      const float* __restrict__ hsc, const float* __restrict__ hsh,
                      const _Float16* __restrict__ w2,     // [16][256] padded
                      const float* __restrict__ b2,
                      float* __restrict__ out) {
  __shared__ __align__(32) _Float16 xl[16][256];           // [col][channel]
  const int tile = blockIdx.x;
  const int lane = threadIdx.x & 31;
  const int wv   = threadIdx.x >> 5;
  const int n  = lane & 15;
  const int hi = lane >> 4;
  const int kbA = hi * 16;
  const int kbB = hi * 16;
  const int gpix = tile * 16 + n;

  // ---- stage 1: wave wv does out-channel tile wv ----
  {
    const _Float16* wrow = w1 + (size_t)(wv * 16 + n) * 256;
    const _Float16* frow = feat + (size_t)gpix * 256;
    v8f acc = {};
    #pragma unroll
    for (int kc = 0; kc < 8; ++kc) {
      v8h alo = *(const v8h*)(wrow + kc * 32 + kbA);
      v8h ahi = *(const v8h*)(wrow + kc * 32 + kbA + 8);
      v16h a;
      #pragma unroll
      for (int i = 0; i < 8; ++i) { a[i] = alo[i]; a[8 + i] = ahi[i]; }
      v16h bf = *(const v16h*)(frow + kc * 32 + kbB);
      acc = __builtin_amdgcn_wmma_f32_16x16x32_f16(false, a, false, bf,
                                                   (short)0, acc, false, false);
    }
    v8h xv;
    #pragma unroll
    for (int r = 0; r < 8; ++r) {
      int o = wv * 16 + 8 * hi + r;
      xv[r] = (_Float16)fmaxf(acc[r] * hsc[o] + hsh[o], 0.0f);
    }
    *(v8h*)(&xl[n][wv * 16 + 8 * hi]) = xv;                // 8 consecutive channels
  }
  __syncthreads();

  // ---- stage 2: wave 0 computes the 10x16 output tile ----
  if (wv == 0) {
    const _Float16* wrow = w2 + (size_t)n * 256;
    v8f acc = {};
    #pragma unroll
    for (int kc = 0; kc < 8; ++kc) {
      v8h alo = *(const v8h*)(wrow + kc * 32 + kbA);
      v8h ahi = *(const v8h*)(wrow + kc * 32 + kbA + 8);
      v16h a;
      #pragma unroll
      for (int i = 0; i < 8; ++i) { a[i] = alo[i]; a[8 + i] = ahi[i]; }
      v16h bf = *(const v16h*)(&xl[n][kc * 32 + kbB]);
      acc = __builtin_amdgcn_wmma_f32_16x16x32_f16(false, a, false, bf,
                                                   (short)0, acc, false, false);
    }
    #pragma unroll
    for (int r = 0; r < 8; ++r) {
      int o = 8 * hi + r;
      if (o < 10) {
        int bt = gpix / 625, p = gpix - bt * 625;
        out[((size_t)(bt * 10 + o)) * 625 + p] = acc[r] + b2[o];
      }
    }
  }
}

// ---------------------------------------------------------------------------
// Host launcher
// ---------------------------------------------------------------------------
static inline size_t alignup256(size_t x) { return (x + 255) & ~(size_t)255; }

extern "C" void kernel_launch(void* const* d_in, const int* in_sizes, int n_in,
                              void* d_out, int out_size, void* d_ws, size_t ws_size,
                              hipStream_t stream) {
  const float* kin  = (const float*)d_in[0];   // [64,256,7,7]
  const float* sin_ = (const float*)d_in[1];   // [64,256,31,31]
  const float* ck_w = (const float*)d_in[2];   // [256,256,3,3]
  const float* ck_g = (const float*)d_in[3];
  const float* ck_b = (const float*)d_in[4];
  const float* ck_m = (const float*)d_in[5];
  const float* ck_v = (const float*)d_in[6];
  const float* cs_w = (const float*)d_in[7];
  const float* cs_g = (const float*)d_in[8];
  const float* cs_b = (const float*)d_in[9];
  const float* cs_m = (const float*)d_in[10];
  const float* cs_v = (const float*)d_in[11];
  const float* h1_w = (const float*)d_in[12];  // [256,256]
  const float* h_g  = (const float*)d_in[13];
  const float* h_b  = (const float*)d_in[14];
  const float* h_m  = (const float*)d_in[15];
  const float* h_v  = (const float*)d_in[16];
  const float* h2_w = (const float*)d_in[17];  // [10,256]
  const float* h2_b = (const float*)d_in[18];
  float* out = (float*)d_out;                  // [64,10,25,25]

  // workspace carve-up
  char* w = (char*)d_ws;
  _Float16* ckw16 = (_Float16*)w; w += alignup256((size_t)256 * 2304 * 2);
  _Float16* csw16 = (_Float16*)w; w += alignup256((size_t)256 * 2304 * 2);
  _Float16* h1w16 = (_Float16*)w; w += alignup256((size_t)256 * 256 * 2);
  _Float16* h2w16 = (_Float16*)w; w += alignup256((size_t)16 * 256 * 2);
  float*    bnbuf = (float*)w;    w += alignup256((size_t)6 * 256 * 4);
  _Float16* kf    = (_Float16*)w; w += alignup256((size_t)64 * 256 * 25 * 2);
  _Float16* sf    = (_Float16*)w; w += alignup256((size_t)64 * 256 * 841 * 2);
  _Float16* featf = (_Float16*)w; w += alignup256((size_t)40000 * 256 * 2);
  float* cksc = bnbuf,        *cksh = bnbuf + 256;
  float* cssc = bnbuf + 512,  *cssh = bnbuf + 768;
  float* hsc  = bnbuf + 1024, *hsh  = bnbuf + 1280;
  (void)ws_size; (void)in_sizes; (void)n_in; (void)out_size;

  // pack weights to f16 + fold BN params
  pack_f16_kernel<<<(256 * 2304 + 255) / 256, 256, 0, stream>>>(ck_w, ckw16, 256 * 2304);
  pack_f16_kernel<<<(256 * 2304 + 255) / 256, 256, 0, stream>>>(cs_w, csw16, 256 * 2304);
  pack_f16_kernel<<<(256 * 256  + 255) / 256, 256, 0, stream>>>(h1_w, h1w16, 256 * 256);
  pack_h2_kernel<<<16, 256, 0, stream>>>(h2_w, h2w16);
  bn_prep_kernel<<<1, 256, 0, stream>>>(ck_g, ck_b, ck_m, ck_v, cksc, cksh);
  bn_prep_kernel<<<1, 256, 0, stream>>>(cs_g, cs_b, cs_m, cs_v, cssc, cssh);
  bn_prep_kernel<<<1, 256, 0, stream>>>(h_g,  h_b,  h_m,  h_v,  hsc,  hsh);

  // branch convs: grid.x = pixel-tile groups (4 tiles/block), grid.y = 4 otile groups
  conv3_bn_relu_wmma<<<dim3(25, 4), 128, 0, stream>>>(
      kin, ckw16, cksc, cksh, kf, 7, 7, 5, 5);
  conv3_bn_relu_wmma<<<dim3(841, 4), 128, 0, stream>>>(
      sin_, csw16, cssc, cssh, sf, 31, 31, 29, 29);

  // depthwise xcorr: one block per (b,c)
  xcorr_dw_kernel<<<64 * 256, 128, 0, stream>>>(sf, kf, featf);

  // fused head: 40000 pixels / 16 = 2500 tiles
  head_wmma_kernel<<<2500, 512, 0, stream>>>(
      featf, h1w16, hsc, hsh, h2w16, h2_b, out);
}
